// InfiniteMultiHeadAttention_44306882625834
// MI455X (gfx1250) — compile-verified, matrix operand order checked by reading, not run on hardware
//
#include <hip/hip_runtime.h>
#include <math.h>

// ---------------------------------------------------------------------------
// CDNA5 / gfx1250 WMMA types
// ---------------------------------------------------------------------------
typedef __attribute__((ext_vector_type(16))) __bf16 v16bf;
typedef __attribute__((ext_vector_type(8)))  float  v8f;

union FragU { unsigned u[8]; v16bf v; };

// ---------------------------------------------------------------------------
// constants
// ---------------------------------------------------------------------------
#define DMODEL 1024
#define NB     8
#define TT     2048
#define NH     16
#define DQ     64
#define SEG    256
#define NCMP   6            // compressed segments
#define TK     518          // NCMP + 2*SEG
#define TKP    544          // TK padded to 17 x 32-column chunks
#define MROWS  (NB*TT)      // 16384

// ---------------------------------------------------------------------------
// helpers
// ---------------------------------------------------------------------------
__device__ __forceinline__ unsigned short bfbits(float f) {
  unsigned u = __builtin_bit_cast(unsigned, f);
  u += 0x7fffu + ((u >> 16) & 1u);          // round-to-nearest-even
  return (unsigned short)(u >> 16);
}
__device__ __forceinline__ unsigned pack2bf(float a, float b) {
  return (unsigned)bfbits(a) | ((unsigned)bfbits(b) << 16);
}
// 16-bit A/B fragment K-index map (ISA 7.12.2): vgpr i, pair at +1
__device__ __forceinline__ int frag_k(int i, int hi8) {
  return ((i & 4) << 2) + hi8 + ((i & 3) << 1);
}
// load one 16x32 bf16 fragment from a row-major [rows, strideK] bf16 matrix
__device__ __forceinline__ v16bf load_frag_bf16(const unsigned short* __restrict__ base,
                                                int strideK, int row, int kk, int hi8) {
  FragU f;
  const unsigned short* rp = base + (size_t)row * strideK + kk;
#pragma unroll
  for (int i = 0; i < 8; ++i)
    f.u[i] = *(const unsigned*)(rp + frag_k(i, hi8));
  return f.v;
}

// ---------------------------------------------------------------------------
// weight prep: fp32 -> bf16, and group-summed Wq
// ---------------------------------------------------------------------------
__global__ void cvt_bf16_kernel(const float* __restrict__ in,
                                unsigned short* __restrict__ out, int n) {
  int i = blockIdx.x * 256 + threadIdx.x;
  if (i < n) out[i] = bfbits(in[i]);
}

__global__ void sumq_kernel(const float* __restrict__ Wq,
                            unsigned short* __restrict__ out) {
  int i = blockIdx.x * 256 + threadIdx.x;          // 1024*1024
  int r = i >> 10, c = i & 1023;
  out[i] = bfbits(Wq[(size_t)r * 1024 + c] + Wq[(size_t)(r + 1024) * 1024 + c]);
}

// ---------------------------------------------------------------------------
// LayerNorm (row of 1024) -> bf16
// ---------------------------------------------------------------------------
__global__ __launch_bounds__(256)
void ln_kernel(const float* __restrict__ q, const float* __restrict__ k,
               const float* __restrict__ v, const float* __restrict__ w,
               const float* __restrict__ b,
               unsigned short* __restrict__ qn, unsigned short* __restrict__ kn,
               unsigned short* __restrict__ vn) {
  int row = blockIdx.x;                       // 0 .. 3*16384-1
  int which = row / MROWS;
  int r = row - which * MROWS;
  const float* in = (which == 0) ? q : (which == 1) ? k : v;
  unsigned short* out = (which == 0) ? qn : (which == 1) ? kn : vn;
  in  += (size_t)r * DMODEL;
  out += (size_t)r * DMODEL;
  int tid = threadIdx.x;
  float x[4], s = 0.f, ss = 0.f;
#pragma unroll
  for (int i = 0; i < 4; ++i) {
    x[i] = in[tid + 256 * i];
    s += x[i];
    ss += x[i] * x[i];
  }
  __shared__ float rs[256], rss[256];
  rs[tid] = s; rss[tid] = ss;
  __syncthreads();
  for (int off = 128; off > 0; off >>= 1) {
    if (tid < off) { rs[tid] += rs[tid + off]; rss[tid] += rss[tid + off]; }
    __syncthreads();
  }
  float mu   = rs[0] * (1.f / DMODEL);
  float var  = rss[0] * (1.f / DMODEL) - mu * mu;
  float rstd = rsqrtf(var + 1e-5f);
#pragma unroll
  for (int i = 0; i < 4; ++i) {
    int c = tid + 256 * i;
    out[c] = bfbits((x[i] - mu) * rstd * w[c] + b[c]);
  }
}

// ---------------------------------------------------------------------------
// bf16 WMMA GEMM:  C[M,N](f32) = A[M,K](bf16,rm) * B[N,K](bf16,rm)^T
// block = 8 waves; block tile 128M x 64N; wave tile 16M x 64N
// K-loop ping-pong unrolled x2: two static buffer sets, each refreshed by
// fresh loads (no register rotation copies), loads overlap WMMAs
// ---------------------------------------------------------------------------
__global__ __launch_bounds__(256)
void gemm_bf16_wmma(const unsigned short* __restrict__ A,
                    const unsigned short* __restrict__ B,
                    float* __restrict__ C, int M, int N, int K) {
  int lane = threadIdx.x & 31;
  int wave = threadIdx.x >> 5;
  int ln   = lane & 15;
  int hi8  = (lane >> 4) << 3;
  int mtile = blockIdx.y * 128 + wave * 16;
  int nbase = blockIdx.x * 64;
  int arow  = mtile + ln;

  v8f acc[4] = {};
  // prologue: buffer set 0 <- chunk 0
  v16bf a0 = load_frag_bf16(A, K, arow, 0, hi8);
  v16bf b0[4];
#pragma unroll
  for (int j = 0; j < 4; ++j)
    b0[j] = load_frag_bf16(B, K, nbase + j * 16 + ln, 0, hi8);

  int kk = 0;
  for (; kk < K - 64; kk += 64) {
    // set 1 <- kk+32 (overlaps set-0 WMMAs)
    __builtin_prefetch(A + (size_t)arow * K + kk + 64, 0, 3);
    v16bf a1 = load_frag_bf16(A, K, arow, kk + 32, hi8);
    v16bf b1[4];
#pragma unroll
    for (int j = 0; j < 4; ++j)
      b1[j] = load_frag_bf16(B, K, nbase + j * 16 + ln, kk + 32, hi8);
#pragma unroll
    for (int j = 0; j < 4; ++j)
      acc[j] = __builtin_amdgcn_wmma_f32_16x16x32_bf16(
                   false, a0, false, b0[j], (short)0, acc[j], false, false);
    // set 0 <- kk+64 (overlaps set-1 WMMAs); fresh loads, no copies
    a0 = load_frag_bf16(A, K, arow, kk + 64, hi8);
#pragma unroll
    for (int j = 0; j < 4; ++j)
      b0[j] = load_frag_bf16(B, K, nbase + j * 16 + ln, kk + 64, hi8);
#pragma unroll
    for (int j = 0; j < 4; ++j)
      acc[j] = __builtin_amdgcn_wmma_f32_16x16x32_bf16(
                   false, a1, false, b1[j], (short)0, acc[j], false, false);
  }
  // tail: chunks kk and kk+32
  {
    v16bf a1 = load_frag_bf16(A, K, arow, kk + 32, hi8);
    v16bf b1[4];
#pragma unroll
    for (int j = 0; j < 4; ++j)
      b1[j] = load_frag_bf16(B, K, nbase + j * 16 + ln, kk + 32, hi8);
#pragma unroll
    for (int j = 0; j < 4; ++j)
      acc[j] = __builtin_amdgcn_wmma_f32_16x16x32_bf16(
                   false, a0, false, b0[j], (short)0, acc[j], false, false);
#pragma unroll
    for (int j = 0; j < 4; ++j)
      acc[j] = __builtin_amdgcn_wmma_f32_16x16x32_bf16(
                   false, a1, false, b1[j], (short)0, acc[j], false, false);
  }

#pragma unroll
  for (int j = 0; j < 4; ++j)
#pragma unroll
    for (int v = 0; v < 8; ++v)
      C[(size_t)(mtile + v + hi8) * N + nbase + j * 16 + ln] = acc[j][v];
}

// ---------------------------------------------------------------------------
// interleaved RoPE, in place, on [N, t, H*64] fp32 (qsum and kh)
// ---------------------------------------------------------------------------
__global__ void rope_kernel(float* __restrict__ qsum, float* __restrict__ kh) {
  float* buf = blockIdx.y ? kh : qsum;
  int idx = blockIdx.x * 256 + threadIdx.x;     // N*t*H*32
  int pi = idx & 31;
  int h  = (idx >> 5) & 15;
  int t  = (idx >> 9) & 2047;
  int n  = idx >> 20;
  float inv = __powf(10000.f, -(float)(2 * pi) * (1.f / DQ));
  float ang = (float)t * inv;
  float c, s;
  __sincosf(ang, &s, &c);
  float* p = buf + ((size_t)(n * TT + t) * DMODEL + h * DQ + 2 * pi);
  float x0 = p[0], x1 = p[1];
  p[0] = x0 * c - x1 * s;
  p[1] = x1 * c + x0 * s;
}

// ---------------------------------------------------------------------------
// segment compression: elu(flatten(seg) @ Wc.T + bc)  -> [N,H,6,64] fp32
// ---------------------------------------------------------------------------
__global__ __launch_bounds__(256)
void compress_kernel(const float* __restrict__ Xk, const float* __restrict__ Xv,
                     const float* __restrict__ Wkc, const float* __restrict__ bkc,
                     const float* __restrict__ Wvc, const float* __restrict__ bvc,
                     float* __restrict__ Ks, float* __restrict__ Vs) {
  int which = blockIdx.y;
  const float* X = which ? Xv : Xk;
  const float* W = which ? Wvc : Wkc;
  const float* bias = which ? bvc : bkc;
  float* out = which ? Vs : Ks;
  int idx = blockIdx.x;                 // 0..767
  int s = idx % NCMP;
  int h = (idx / NCMP) & 15;
  int n = idx / (NCMP * NH);
  int tid = threadIdx.x;
  int dq = tid & 63, part = tid >> 6;   // 4 j-chunks of 64
  float acc = 0.f;
  for (int j = part * 64; j < part * 64 + 64; ++j) {
    const float4* xr = (const float4*)(X + ((size_t)(n * TT + s * SEG + j) * DMODEL + h * DQ));
    const float4* wr = (const float4*)(W + (size_t)dq * (SEG * DQ) + j * DQ);
#pragma unroll
    for (int d = 0; d < 16; ++d) {
      float4 a = xr[d], w4 = wr[d];
      acc += a.x * w4.x + a.y * w4.y + a.z * w4.z + a.w * w4.w;
    }
  }
  __shared__ float red[256];
  red[tid] = acc;
  __syncthreads();
  if (part == 0) {
    float t = red[dq] + red[dq + 64] + red[dq + 128] + red[dq + 192] + bias[dq];
    float e = (t > 0.f) ? t : (__expf(t) - 1.f);
    out[((size_t)(n * NH + h) * NCMP + s) * DQ + dq] = e;
  }
}

// ---------------------------------------------------------------------------
// pack concat-K / transposed-V into bf16 tiles for the flash loop
//   Kp[N,H,TKP,64]  : row T = {compressed 0..5 | recent 6..517 | zero pad}
//   Vt[N,H,64,TKP]  : same, transposed so V B-frags are contiguous along T
// ---------------------------------------------------------------------------
__global__ void pack_kv_kernel(const float* __restrict__ Kh, const float* __restrict__ Vh,
                               const float* __restrict__ Ks, const float* __restrict__ Vs,
                               unsigned short* __restrict__ Kp,
                               unsigned short* __restrict__ Vt) {
  int idx = blockIdx.x * 256 + threadIdx.x;   // N*H*TKP*64
  int dq = idx & 63;
  int T  = (idx >> 6) % TKP;
  int nh = idx / (TKP * 64);
  int n = nh >> 4, h = nh & 15;
  float kv = 0.f, vv = 0.f;
  if (T < NCMP) {
    kv = Ks[((size_t)nh * NCMP + T) * DQ + dq];
    vv = Vs[((size_t)nh * NCMP + T) * DQ + dq];
  } else if (T < TK) {
    size_t src = (size_t)(n * TT + (TT - 2 * SEG) + T - NCMP) * DMODEL + h * DQ + dq;
    kv = Kh[src];
    vv = Vh[src];
  }
  Kp[((size_t)nh * TKP + T) * 64 + dq] = bfbits(kv);
  Vt[((size_t)nh * 64 + dq) * TKP + T] = bfbits(vv);
}

// ---------------------------------------------------------------------------
// fused flash attention (bf16 WMMA, online softmax, causal, tanh-capped)
// grid: N*H*16 blocks; 8 waves/block, each wave = one 16-row t tile
// ---------------------------------------------------------------------------
__global__ __launch_bounds__(256)
void attn_kernel(const float* __restrict__ Q,          // [N,t,H*64] roped, g-summed
                 const unsigned short* __restrict__ Kp, // bf16 [N,H,TKP,64]
                 const unsigned short* __restrict__ Vt, // bf16 [N,H,64,TKP]
                 unsigned short* __restrict__ O)        // bf16 [N,t,H*64]
{
  __shared__ unsigned short pshm[8][16 * 32];
  int lane = threadIdx.x & 31;
  int wave = threadIdx.x >> 5;
  int ln   = lane & 15;
  int hi8  = (lane >> 4) << 3;
  int nh    = blockIdx.x >> 4;
  int chunk = blockIdx.x & 15;
  int n = nh >> 4, h = nh & 15;
  int tbase = (chunk * 8 + wave) * 16;
  int kbase = nh * TKP;        // row base into Kp
  int vbase = nh * 64;         // row base into Vt

  // ---- Q fragments (dq 0..31 / 32..63), fp32 -> bf16 once
  const float* qrow = Q + ((size_t)(n * TT + tbase + ln) * DMODEL + h * DQ);
  FragU qf0, qf1;
#pragma unroll
  for (int i = 0; i < 8; ++i) {
    int k0 = frag_k(i, hi8);
    qf0.u[i] = pack2bf(qrow[k0],      qrow[k0 + 1]);
    qf1.u[i] = pack2bf(qrow[32 + k0], qrow[32 + k0 + 1]);
  }

  float m_i[8], l_i[8];
  v8f acc[4] = {};
#pragma unroll
  for (int v = 0; v < 8; ++v) { m_i[v] = -1e30f; l_i[v] = 0.f; }

  int nT = min(TK, chunk * 128 + 128);     // uniform per block -> barrier-safe
  int nIter = (nT + 31) >> 5;

  for (int it = 0; it < nIter; ++it) {
    int Tb = it * 32;

    // ---- scores S[16 x 32] = q @ K^T  (two 16-col WMMA tiles)
    v8f s[2] = {};
#pragma unroll
    for (int tt = 0; tt < 2; ++tt) {
      int krow = kbase + Tb + tt * 16 + ln;
      v16bf b0 = load_frag_bf16(Kp, 64, krow, 0,  hi8);
      v16bf b1 = load_frag_bf16(Kp, 64, krow, 32, hi8);
      s[tt] = __builtin_amdgcn_wmma_f32_16x16x32_bf16(
                  false, qf0.v, false, b0, (short)0, s[tt], false, false);
      s[tt] = __builtin_amdgcn_wmma_f32_16x16x32_bf16(
                  false, qf1.v, false, b1, (short)0, s[tt], false, false);
    }

    // ---- scale, tanh cap, causal + pad mask (C-layout: row = v+hi8, col = ln)
    float sc[2][8];
#pragma unroll
    for (int tt = 0; tt < 2; ++tt) {
      int Tcol = Tb + tt * 16 + ln;
#pragma unroll
      for (int v = 0; v < 8; ++v) {
        int trow = tbase + v + hi8;
        float x = s[tt][v] * 0.125f;               // 1/sqrt(64)
        x = 30.f * tanhf(x * (1.f / 30.f));
        bool ok = (Tcol < TK) && (Tcol <= trow);
        sc[tt][v] = ok ? x : -1e30f;
      }
    }

    // ---- row max over 16 lanes of each half-wave
    float tm[8];
#pragma unroll
    for (int v = 0; v < 8; ++v) tm[v] = fmaxf(sc[0][v], sc[1][v]);
#pragma unroll
    for (int off = 8; off >= 1; off >>= 1)
#pragma unroll
      for (int v = 0; v < 8; ++v)
        tm[v] = fmaxf(tm[v], __shfl_xor(tm[v], off, 16));

    // ---- online softmax update
    float alpha[8], rs[8];
#pragma unroll
    for (int v = 0; v < 8; ++v) {
      float mn = fmaxf(m_i[v], tm[v]);
      alpha[v] = __expf(m_i[v] - mn);
      m_i[v] = mn;
      rs[v] = 0.f;
    }
#pragma unroll
    for (int tt = 0; tt < 2; ++tt)
#pragma unroll
      for (int v = 0; v < 8; ++v) {
        float p = __expf(sc[tt][v] - m_i[v]);
        sc[tt][v] = p;
        rs[v] += p;
      }
#pragma unroll
    for (int off = 8; off >= 1; off >>= 1)
#pragma unroll
      for (int v = 0; v < 8; ++v) rs[v] += __shfl_xor(rs[v], off, 16);
#pragma unroll
    for (int v = 0; v < 8; ++v) l_i[v] = l_i[v] * alpha[v] + rs[v];
#pragma unroll
    for (int d = 0; d < 4; ++d)
#pragma unroll
      for (int v = 0; v < 8; ++v) acc[d][v] *= alpha[v];

    // ---- P: C-layout -> A-fragment via LDS (per-wave 16x32 bf16 tile)
    __syncthreads();
#pragma unroll
    for (int tt = 0; tt < 2; ++tt)
#pragma unroll
      for (int v = 0; v < 8; ++v)
        pshm[wave][(v + hi8) * 32 + tt * 16 + ln] = bfbits(sc[tt][v]);
    __syncthreads();
    FragU pa;
#pragma unroll
    for (int i = 0; i < 8; ++i)
      pa.u[i] = *(const unsigned*)&pshm[wave][ln * 32 + frag_k(i, hi8)];

    // ---- O += P @ V  (4 dv tiles; Vt rows are contiguous along T)
#pragma unroll
    for (int d = 0; d < 4; ++d) {
      v16bf vb = load_frag_bf16(Vt, TKP, vbase + d * 16 + ln, Tb, hi8);
      acc[d] = __builtin_amdgcn_wmma_f32_16x16x32_bf16(
                   false, pa.v, false, vb, (short)0, acc[d], false, false);
    }
  }

  // ---- epilogue: normalize and store bf16 [n, t, h*64+dv]
#pragma unroll
  for (int d = 0; d < 4; ++d)
#pragma unroll
    for (int v = 0; v < 8; ++v) {
      float o = acc[d][v] / l_i[v];
      O[(size_t)(n * TT + tbase + v + hi8) * DMODEL + h * DQ + d * 16 + ln] = bfbits(o);
    }
}

// ---------------------------------------------------------------------------
// host-side launcher
// ---------------------------------------------------------------------------
extern "C" void kernel_launch(void* const* d_in, const int* in_sizes, int n_in,
                              void* d_out, int out_size, void* d_ws, size_t ws_size,
                              hipStream_t stream) {
  const float* qseq = (const float*)d_in[0];
  const float* kseq = (const float*)d_in[1];
  const float* vseq = (const float*)d_in[2];
  const float* lnw  = (const float*)d_in[3];
  const float* lnb  = (const float*)d_in[4];
  const float* Wq   = (const float*)d_in[5];
  const float* Wk   = (const float*)d_in[6];
  const float* Wv   = (const float*)d_in[7];
  const float* Wo   = (const float*)d_in[8];
  const float* Wkc  = (const float*)d_in[9];
  const float* bkc  = (const float*)d_in[10];
  const float* Wvc  = (const float*)d_in[11];
  const float* bvc  = (const float*)d_in[12];

  char* ws = (char*)d_ws;
  size_t off = 0;
  auto take = [&](size_t bytes) -> void* {
    void* p = ws + off;
    off = (off + bytes + 255) & ~(size_t)255;
    return p;
  };
  const size_t actB = (size_t)MROWS * DMODEL;      // 16.7M elems
  unsigned short* qn   = (unsigned short*)take(actB * 2);
  unsigned short* kn   = (unsigned short*)take(actB * 2);
  unsigned short* vn   = (unsigned short*)take(actB * 2);
  unsigned short* Wqs  = (unsigned short*)take((size_t)1024 * 1024 * 2);
  unsigned short* Wkb  = (unsigned short*)take((size_t)1024 * 1024 * 2);
  unsigned short* Wvb  = (unsigned short*)take((size_t)1024 * 1024 * 2);
  unsigned short* Wob  = (unsigned short*)take((size_t)1024 * 1024 * 2);
  float* qsum = (float*)take(actB * 4);
  float* khb  = (float*)take(actB * 4);
  float* vhb  = (float*)take(actB * 4);
  float* ksum = (float*)take((size_t)NB * NH * NCMP * DQ * 4);
  float* vsum = (float*)take((size_t)NB * NH * NCMP * DQ * 4);
  unsigned short* Kp   = (unsigned short*)take((size_t)NB * NH * TKP * 64 * 2);
  unsigned short* Vt   = (unsigned short*)take((size_t)NB * NH * 64 * TKP * 2);
  unsigned short* obuf = (unsigned short*)take(actB * 2);

  // 1) weight prep
  cvt_bf16_kernel<<<4096, 256, 0, stream>>>(Wk, Wkb, 1024 * 1024);
  cvt_bf16_kernel<<<4096, 256, 0, stream>>>(Wv, Wvb, 1024 * 1024);
  cvt_bf16_kernel<<<4096, 256, 0, stream>>>(Wo, Wob, 1024 * 1024);
  sumq_kernel<<<4096, 256, 0, stream>>>(Wq, Wqs);

  // 2) layernorm -> bf16
  ln_kernel<<<3 * MROWS, 256, 0, stream>>>(qseq, kseq, vseq, lnw, lnb, qn, kn, vn);

  // 3) projections (bf16 WMMA)
  dim3 ggrid(1024 / 64, MROWS / 128);
  gemm_bf16_wmma<<<ggrid, 256, 0, stream>>>(qn, Wqs, qsum, MROWS, 1024, 1024);
  gemm_bf16_wmma<<<ggrid, 256, 0, stream>>>(kn, Wkb, khb,  MROWS, 1024, 1024);
  gemm_bf16_wmma<<<ggrid, 256, 0, stream>>>(vn, Wvb, vhb,  MROWS, 1024, 1024);

  // 4) RoPE (q summed over groups first -- RoPE is linear, so this commutes)
  rope_kernel<<<dim3((NB * TT * NH * 32) / 256, 2), 256, 0, stream>>>(qsum, khb);

  // 5) segment compression
  compress_kernel<<<dim3(NB * NH * NCMP, 2), 256, 0, stream>>>(
      khb, vhb, Wkc, bkc, Wvc, bvc, ksum, vsum);

  // 6) pack concat-K (bf16) and transposed-V (bf16) for the flash loop
  pack_kv_kernel<<<(NB * NH * TKP * 64) / 256, 256, 0, stream>>>(
      khb, vhb, ksum, vsum, Kp, Vt);

  // 7) fused flash attention -> bf16 O
  attn_kernel<<<NB * NH * 16, 256, 0, stream>>>(qsum, Kp, Vt, obuf);

  // 8) output projection -> fp32 d_out
  gemm_bf16_wmma<<<ggrid, 256, 0, stream>>>(obuf, Wob, (float*)d_out,
                                            MROWS, 1024, 1024);
}
